// QuantizedLinear_31748398252443
// MI455X (gfx1250) — 8-bit matrix operand order repaired and checked by reading, compiled, not run on hardware
//
#include <hip/hip_runtime.h>
#include <cstdint>
#include <cstddef>


typedef __attribute__((ext_vector_type(8))) int v8i;
typedef int v4i_b __attribute__((vector_size(16)));

#define BM 128
#define BN 128
#define BK 64
#define A_BYTES (BM * BK)
#define B_BYTES (BN * BK)
#define STAGE_BYTES (A_BYTES + B_BYTES)
#define STAGE_CHUNKS (STAGE_BYTES / 16)

#if defined(__AMDGCN__) && __has_builtin(__builtin_amdgcn_global_load_async_to_lds_b128)
#define USE_ASYNC_LDS 1
typedef __attribute__((address_space(1))) v4i_b* async_gptr_t;
typedef __attribute__((address_space(3))) v4i_b* async_lptr_t;
#else
#define USE_ASYNC_LDS 0
#endif

__global__ __launch_bounds__(256) void quant_act_kernel(
    const float* __restrict__ x, const float* __restrict__ act_scale_p,
    const int* __restrict__ act_zp_p, int8_t* __restrict__ xq8, int n4) {
  int i = blockIdx.x * blockDim.x + threadIdx.x;
  if (i >= n4) return;
  float inv = 1.0f / (*act_scale_p);
  int azp = *act_zp_p;
  float lo = -(float)azp;
  float hi = 255.0f - (float)azp;
  float4 v = reinterpret_cast<const float4*>(x)[i];
  int a = (int)fminf(fmaxf(rintf(v.x * inv), lo), hi);
  int b = (int)fminf(fmaxf(rintf(v.y * inv), lo), hi);
  int c = (int)fminf(fmaxf(rintf(v.z * inv), lo), hi);
  int d = (int)fminf(fmaxf(rintf(v.w * inv), lo), hi);
  int packed = (a & 0xFF) | ((b & 0xFF) << 8) | ((c & 0xFF) << 16) | ((d & 0xFF) << 24);
  reinterpret_cast<int*>(xq8)[i] = packed;
}

__global__ __launch_bounds__(256) void pack_w_kernel(
    const int* __restrict__ qw, const int* __restrict__ zp_p,
    int8_t* __restrict__ w8, int n4) {
  int i = blockIdx.x * blockDim.x + threadIdx.x;
  if (i >= n4) return;
  int zp = *zp_p;
  int4 q = reinterpret_cast<const int4*>(qw)[i];
  int a = q.x - zp, b = q.y - zp, c = q.z - zp, d = q.w - zp;
  int packed = (a & 0xFF) | ((b & 0xFF) << 8) | ((c & 0xFF) << 16) | ((d & 0xFF) << 24);
  reinterpret_cast<int*>(w8)[i] = packed;
}

__device__ inline v8i load_a_frag(const char* Ab, int mrow, int half) {
  const char* base = Ab + mrow * BK + half * 16;
  int4 d0 = *reinterpret_cast<const int4*>(base + 0);
  int4 d1 = *reinterpret_cast<const int4*>(base + 32);
  v8i a;
  a[0] = d0.x; a[1] = d0.y; a[2] = d0.z; a[3] = d0.w;
  a[4] = d1.x; a[5] = d1.y; a[6] = d1.z; a[7] = d1.w;
  return a;
}

__device__ inline v8i load_b_frag(const char* Bb, int ncol, int half) {
  const char* base = Bb + ncol * BK + half * 16;
  int4 d0 = *reinterpret_cast<const int4*>(base + 0);
  int4 d1 = *reinterpret_cast<const int4*>(base + 32);
  v8i b;
  b[0] = d0.x; b[1] = d0.y; b[2] = d0.z; b[3] = d0.w;
  b[4] = d1.x; b[5] = d1.y; b[6] = d1.z; b[7] = d1.w;
  return b;
}

__device__ inline void stage_tile(const int8_t* __restrict__ A,
                                  const int8_t* __restrict__ W,
                                  int4* lds_stage, int m_base, int n_base,
                                  int K, int k0, int tid) {
  #pragma unroll
  for (int t = 0; t < 4; ++t) {
    int c = tid + t * 256;
    const int8_t* g;
    if (c < A_BYTES / 16) {
      int row = c >> 2, coff = (c & 3) * 16;
      g = A + (size_t)(m_base + row) * K + k0 + coff;
    } else {
      int c2 = c - A_BYTES / 16;
      int row = c2 >> 2, coff = (c2 & 3) * 16;
      g = W + (size_t)(n_base + row) * K + k0 + coff;
    }
#if USE_ASYNC_LDS
    __builtin_amdgcn_global_load_async_to_lds_b128(
        (async_gptr_t)(int8_t*)g,
        (async_lptr_t)&lds_stage[c],
        0, 0);
#else
    lds_stage[c] = *reinterpret_cast<const int4*>(g);
#endif
  }
}

__device__ inline void wait_stage() {
#if USE_ASYNC_LDS
#if __has_builtin(__builtin_amdgcn_s_wait_asynccnt)
  __builtin_amdgcn_s_wait_asynccnt(0);
#else
  asm volatile("s_wait_asynccnt 0x0" ::: "memory");
#endif
#endif
}

__global__ __launch_bounds__(256) void gemm_i8_kernel(
    const int8_t* __restrict__ A,
    const int8_t* __restrict__ W,
    const float* __restrict__ bias,
    const float* __restrict__ scale_p,
    const float* __restrict__ act_scale_p,
    float* __restrict__ out,
    int M, int N, int K) {
  __shared__ int4 lds[2 * STAGE_CHUNKS];

  const int tid  = threadIdx.x;
  const int lane = tid & 31;
  const int wave = tid >> 5;
  const int half = lane >> 4;
  const int l16  = lane & 15;
  const int wm   = wave >> 1;
  const int wn   = wave & 1;

  const int m_base = blockIdx.y * BM;
  const int n_base = blockIdx.x * BN;

  v8i acc[2][4] = {};
  const int nsteps = K / BK;

  stage_tile(A, W, lds, m_base, n_base, K, 0, tid);

  for (int i = 0; i < nsteps; ++i) {
    wait_stage();
    __syncthreads();
    if (i + 1 < nsteps)
      stage_tile(A, W, lds + ((i + 1) & 1) * STAGE_CHUNKS,
                 m_base, n_base, K, (i + 1) * BK, tid);

    const char* base = reinterpret_cast<const char*>(lds) + (i & 1) * STAGE_BYTES;
    const char* Ab = base;
    const char* Bb = base + A_BYTES;

    v8i a0 = load_a_frag(Ab, wm * 32 + 0  + l16, half);
    v8i a1 = load_a_frag(Ab, wm * 32 + 16 + l16, half);
    v8i b[4];
    #pragma unroll
    for (int j = 0; j < 4; ++j)
      b[j] = load_b_frag(Bb, wn * 64 + j * 16 + l16, half);

    #pragma unroll
    for (int j = 0; j < 4; ++j)
      acc[0][j] = __builtin_amdgcn_wmma_i32_16x16x64_iu8(
          true, a0, true, b[j], acc[0][j], false, false);
    #pragma unroll
    for (int j = 0; j < 4; ++j)
      acc[1][j] = __builtin_amdgcn_wmma_i32_16x16x64_iu8(
          true, a1, true, b[j], acc[1][j], false, false);
  }

  const float cs = (*scale_p) * (*act_scale_p);
  #pragma unroll
  for (int i = 0; i < 2; ++i) {
    #pragma unroll
    for (int j = 0; j < 4; ++j) {
      int n_g = n_base + wn * 64 + j * 16 + l16;
      float bv = bias[n_g];
      #pragma unroll
      for (int r = 0; r < 8; ++r) {
        int m_g = m_base + wm * 32 + i * 16 + half * 8 + r;
        out[(size_t)m_g * N + n_g] = (float)acc[i][j][r] * cs + bv;
      }
    }
  }
}

extern "C" void kernel_launch(void* const* d_in, const int* in_sizes, int n_in,
                              void* d_out, int out_size, void* d_ws, size_t ws_size,
                              hipStream_t stream) {
  const float* x          = (const float*)d_in[0];
  const int*   qweight    = (const int*)d_in[1];
  const float* scale      = (const float*)d_in[2];
  const int*   zero_point = (const int*)d_in[3];
  const float* bias       = (const float*)d_in[4];
  const float* act_scale  = (const float*)d_in[5];
  const int*   act_zp     = (const int*)d_in[6];
  float* out = (float*)d_out;

  const int    N = in_sizes[4];
  const long   K = (long)in_sizes[1] / N;
  const long   M = (long)in_sizes[0] / K;

  int8_t* xq8 = reinterpret_cast<int8_t*>(d_ws);
  int8_t* w8  = xq8 + (size_t)M * K;

  {
    int n4 = (int)(M * K / 4);
    quant_act_kernel<<<(n4 + 255) / 256, 256, 0, stream>>>(x, act_scale, act_zp, xq8, n4);
  }
  {
    int n4 = (int)((size_t)N * K / 4);
    pack_w_kernel<<<(n4 + 255) / 256, 256, 0, stream>>>(qweight, zero_point, w8, n4);
  }
  {
    dim3 grid((unsigned)(N / BN), (unsigned)(M / BM));
    gemm_i8_kernel<<<grid, 256, 0, stream>>>(xq8, w8, bias, scale, act_scale,
                                             out, (int)M, N, (int)K);
  }
}
